// Model_16312285790557
// MI455X (gfx1250) — matrix operand order repaired and checked by reading, not compiled, not run on hardware
//
#include <hip/hip_runtime.h>
#include <hip/hip_bf16.h>
#include <hip/hip_fp16.h>

typedef __attribute__((ext_vector_type(16))) _Float16 v16h;
typedef __attribute__((ext_vector_type(8)))  float    v8f;
typedef __attribute__((ext_vector_type(4)))  unsigned v4u;

// ---------------------------------------------------------------------------
// Fragment layouts (wave32, CDNA5 ISA 7.12.2):
//  A-frag (weights):  flat = ((tm*KC + kc)*32 + lane)*16 + e
//     m = tm*16 + (lane&15),  k = kc*32 + (lane&16) + e
//  B-frag (acts):     flat = ((kc*Mtiles + tn)*32 + lane)*16 + e
//     k = kc*32 + (lane&16) + e,  n = tn*16 + (lane&15)
//  Each lane's 16 halves are 32 contiguous bytes -> two b128 loads.
//  D results remap into B-frag of the next layer with 8 contiguous halves
//  per lane -> one b128 store.
// ---------------------------------------------------------------------------

__device__ __forceinline__ v16h load_frag16(const _Float16* p) {
  union { v16h h; v4u u[2]; } r;
  r.u[0] = *(const v4u*)p;
  r.u[1] = *(const v4u*)(p + 8);
  return r.h;
}

// ---------------------------------------------------------------------------
// Pack f32 weights (Cout x K) into A-fragment f16, zero-padded.
// ---------------------------------------------------------------------------
__global__ void pack_w_kernel(const float* __restrict__ W, _Float16* __restrict__ Ap,
                              int Cout, int K, int KC, int total)
{
  int t = blockIdx.x * blockDim.x + threadIdx.x;
  if (t >= total) return;
  int e = t & 15, lane = (t >> 4) & 31, tile = t >> 9;
  int kc = tile % KC, tm = tile / KC;
  int m = tm * 16 + (lane & 15);
  int akb = (lane < 16) ? 0 : 8;
  int ka = kc * 32 + (lane & 16) + e;
  float w = (m < Cout && ka < K) ? W[(size_t)m * K + ka] : 0.0f;
  Ap[t] = (_Float16)w;
}

// ---------------------------------------------------------------------------
// WMMA GEMM, fully fragment-resident.  One wave per 16x16 output tile.
//   Ap: A-frag weights, G: B-frag input, O: B-frag output (next layer's input)
// ---------------------------------------------------------------------------
__global__ __launch_bounds__(32) void wmma_gemm_kernel(
    const _Float16* __restrict__ Ap, const float* __restrict__ bias,
    const _Float16* __restrict__ G, _Float16* __restrict__ O,
    int Cout, int KC, int Mtiles, int relu)
{
  const int lane = threadIdx.x;
  const int tn = blockIdx.x, tm = blockIdx.y;

  const _Float16* __restrict__ ap = Ap + ((size_t)tm * KC) * 512 + lane * 16;
  const _Float16* __restrict__ bp = G + ((size_t)tn) * 512 + lane * 16;
  const size_t bstep = (size_t)Mtiles * 512;

  v8f acc = {};
  for (int kc = 0; kc < KC; ++kc) {
    v16h a = load_frag16(ap);
    v16h b = load_frag16(bp);
    if (kc + 1 < KC) __builtin_prefetch(bp + bstep, 0, 1);
    ap += 512;
    bp += bstep;
    acc = __builtin_amdgcn_wmma_f32_16x16x32_f16(
        /*neg_a=*/false, a, /*neg_b=*/false, b,
        /*c_mod=*/(short)0, acc, /*reuse_a=*/false, /*reuse_b=*/false);
  }

  // D row for VGPR v: m = tm*16 + ((lane<16)?0:8) + v
  const int msub = (lane < 16) ? 0 : 8;
  const int mbase = tm * 16 + msub;
  // Output B-frag coords: kc'=tm>>1, tn'=tn, lane'=(lane&15)+((tm&1)<<4), e'=msub+v..(+7)
  const size_t obase =
      ((((size_t)(tm >> 1)) * Mtiles + tn) * 32 + (lane & 15) + ((size_t)(tm & 1) << 4)) * 16 + msub;

  if (mbase + 7 < Cout) {
    union { _Float16 h[8]; v4u u; } st;
#pragma unroll
    for (int v = 0; v < 8; ++v) {
      float r = acc[v] + bias[mbase + v];
      if (relu) r = fmaxf(r, 0.0f);
      st.h[v] = (_Float16)r;
    }
    *(v4u*)(O + obase) = st.u;       // one b128 store per lane
  } else {
#pragma unroll
    for (int v = 0; v < 8; ++v) {
      int m = mbase + v;
      if (m < Cout) {
        float r = acc[v] + bias[m];
        if (relu) r = fmaxf(r, 0.0f);
        O[obase + v] = (_Float16)r;
      }
    }
  }
}

// ---------------------------------------------------------------------------
// Split x (B,N,4) -> xyz0 (B,N,3) f32 and feat0 (B,1,N) f16
// ---------------------------------------------------------------------------
__global__ void split_x_kernel(const float* __restrict__ x, float* __restrict__ xyz0,
                               _Float16* __restrict__ feat0, int total)
{
  int i = blockIdx.x * blockDim.x + threadIdx.x;
  if (i >= total) return;
  xyz0[i * 3 + 0] = x[i * 4 + 0];
  xyz0[i * 3 + 1] = x[i * 4 + 1];
  xyz0[i * 3 + 2] = x[i * 4 + 2];
  feat0[i] = (_Float16)x[i * 4 + 3];
}

// ---------------------------------------------------------------------------
// Farthest point sampling: one block per batch, dist in LDS, LDS tree-reduction argmax
// with lowest-index tie-break (matches jnp.argmax).
// ---------------------------------------------------------------------------
#define FPS_BLOCK 256
__global__ __launch_bounds__(FPS_BLOCK) void fps_kernel(
    const float* __restrict__ xyz, int* __restrict__ fidx, int n, int npoint)
{
  const int b = blockIdx.x;
  const float* X = xyz + (size_t)b * n * 3;
  extern __shared__ char smem[];
  float* dist = (float*)smem;
  float* rmax = (float*)(smem + (size_t)n * 4);
  int*   rarg = (int*)(smem + (size_t)n * 4 + FPS_BLOCK * 4);
  __shared__ int   s_last;
  __shared__ float s_px, s_py, s_pz;

  const int tid = threadIdx.x;
  for (int i = tid; i < n; i += FPS_BLOCK) dist[i] = 1e10f;
  if (tid == 0) { fidx[(size_t)b * npoint] = 0; s_last = 0; }
  __syncthreads();

  for (int it = 1; it < npoint; ++it) {
    if (tid == 0) {
      int l = s_last;
      s_px = X[l * 3 + 0]; s_py = X[l * 3 + 1]; s_pz = X[l * 3 + 2];
    }
    __syncthreads();
    float px = s_px, py = s_py, pz = s_pz;
    float best = -1.0f; int barg = 0;
    for (int i = tid; i < n; i += FPS_BLOCK) {
      float dx = X[i * 3 + 0] - px, dy = X[i * 3 + 1] - py, dz = X[i * 3 + 2] - pz;
      float d = fminf(dist[i], dx * dx + dy * dy + dz * dz);
      dist[i] = d;
      if (d > best) { best = d; barg = i; }
    }
    rmax[tid] = best; rarg[tid] = barg;
    __syncthreads();
    for (int s = FPS_BLOCK / 2; s > 0; s >>= 1) {
      if (tid < s) {
        float o = rmax[tid + s]; int oa = rarg[tid + s];
        if (o > rmax[tid] || (o == rmax[tid] && oa < rarg[tid])) {
          rmax[tid] = o; rarg[tid] = oa;
        }
      }
      __syncthreads();
    }
    if (tid == 0) { s_last = rarg[0]; fidx[(size_t)b * npoint + it] = rarg[0]; }
    __syncthreads();
  }
}

__global__ void gather_xyz_kernel(const float* __restrict__ xyz, const int* __restrict__ fidx,
                                  float* __restrict__ nxyz, int n, int npoint, int total)
{
  int t = blockIdx.x * blockDim.x + threadIdx.x;
  if (t >= total) return;
  int d = t % 3;
  int j = (t / 3) % npoint;
  int b = t / (3 * npoint);
  int p = fidx[(size_t)b * npoint + j];
  nxyz[t] = xyz[((size_t)b * n + p) * 3 + d];
}

// ---------------------------------------------------------------------------
// Ball query: one wave32 per center; first nsample in-radius indices in
// increasing order; pad with first hit (or 0 if none).
// ---------------------------------------------------------------------------
__global__ __launch_bounds__(256) void ballquery_kernel(
    const float* __restrict__ xyz, const float* __restrict__ new_xyz,
    int* __restrict__ idx, int n, int npoint, float r2, int nsample)
{
  const int wavesPerBlock = blockDim.x >> 5;
  const int gw = blockIdx.x * wavesPerBlock + (threadIdx.x >> 5);
  const int lane = threadIdx.x & 31;
  const int b = gw / npoint, j = gw % npoint;
  const float* X = xyz + (size_t)b * n * 3;
  const float cx = new_xyz[((size_t)b * npoint + j) * 3 + 0];
  const float cy = new_xyz[((size_t)b * npoint + j) * 3 + 1];
  const float cz = new_xyz[((size_t)b * npoint + j) * 3 + 2];
  int* out = idx + ((size_t)b * npoint + j) * nsample;

  int count = 0, first = 0;
  for (int base = 0; base < n && count < nsample; base += 32) {
    int i = base + lane;
    float dx = X[i * 3 + 0] - cx, dy = X[i * 3 + 1] - cy, dz = X[i * 3 + 2] - cz;
    bool in = (dx * dx + dy * dy + dz * dz) < r2;
#if __has_builtin(__builtin_amdgcn_ballot_w32)
    unsigned mask = __builtin_amdgcn_ballot_w32(in);
#else
    unsigned mask = (unsigned)__ballot(in);
#endif
    while (mask && count < nsample) {
      int bit = __builtin_ctz(mask);
      mask &= mask - 1;
      if (count == 0) first = base + bit;
      if (lane == 0) out[count] = base + bit;
      ++count;
    }
  }
  for (int s = count + lane; s < nsample; s += 32) out[s] = first;
}

// ---------------------------------------------------------------------------
// Build grouped tensor directly in B-fragment layout (one batch):
//   k<3: xyz[idx]-new_xyz, 3<=k<3+C: feat[k-3, idx], else 0.
// ---------------------------------------------------------------------------
__global__ void group_kernel(const float* __restrict__ xyz, const float* __restrict__ nxyz,
                             const _Float16* __restrict__ feat, const int* __restrict__ idx,
                             _Float16* __restrict__ G,
                             int n, int nsample, int C, int Mtiles, int total)
{
  int t = blockIdx.x * blockDim.x + threadIdx.x;
  if (t >= total) return;
  int e = t & 15, lane = (t >> 4) & 31, tile = t >> 9;
  int tn = tile % Mtiles, kc = tile / Mtiles;
  int k = kc * 32 + (lane & 16) + e;
  int m = tn * 16 + (lane & 15);
  _Float16 val = (_Float16)0.0f;
  if (k < 3 + C) {
    int j = m / nsample;
    int p = idx[(size_t)j * nsample + (m - j * nsample)];
    if (k < 3) val = (_Float16)(xyz[(size_t)p * 3 + k] - nxyz[(size_t)j * 3 + k]);
    else       val = feat[(size_t)(k - 3) * n + p];
  }
  G[t] = val;
}

// Global-group (SA4): concat(raw xyz^T, feat) over 64 points, B-frag layout.
__global__ void group_all_kernel(const float* __restrict__ xyz, const _Float16* __restrict__ feat,
                                 _Float16* __restrict__ G, int M, int C, int Mtiles, int total)
{
  int t = blockIdx.x * blockDim.x + threadIdx.x;
  if (t >= total) return;
  int e = t & 15, lane = (t >> 4) & 31, tile = t >> 9;
  int tn = tile % Mtiles, kc = tile / Mtiles;
  int k = kc * 32 + (lane & 16) + e;
  int m = tn * 16 + (lane & 15);
  _Float16 val = (_Float16)0.0f;
  if (k < 3)          val = (_Float16)xyz[(size_t)m * 3 + k];
  else if (k < 3 + C) val = feat[(size_t)(k - 3) * M + m];
  G[t] = val;
}

// ---------------------------------------------------------------------------
// Max pool over nsample from B-frag activation -> plain [C][npoint] f16.
// ---------------------------------------------------------------------------
__device__ __forceinline__ size_t frag_addr(int k, int m, int Mtiles) {
  return (((size_t)(k >> 5) * Mtiles + (m >> 4)) * 32 + (m & 15) + (((k >> 4) & 1) << 4)) * 16
         + (k & 15);
}

__global__ void maxpool_kernel(const _Float16* __restrict__ A, _Float16* __restrict__ out,
                               int C, int npoint, int nsample, int MtilesIn,
                               int obase, int ostride)
{
  int t = blockIdx.x * blockDim.x + threadIdx.x;
  if (t >= C * npoint) return;
  int c = t / npoint, j = t - c * npoint;
  float mx = -1e30f;
  for (int s = 0; s < nsample; ++s) {
    int m = j * nsample + s;
    mx = fmaxf(mx, (float)A[frag_addr(c, m, MtilesIn)]);
  }
  out[(size_t)obase + (size_t)c * ostride + j] = (_Float16)mx;
}

// SA4 pool: B-frag activation [C x 64] -> featH B-frag column bcol (Mtiles=1).
__global__ void maxpool_frag_kernel(const _Float16* __restrict__ A, _Float16* __restrict__ outF,
                                    int C, int nsample, int MtilesIn, int bcol)
{
  int c = blockIdx.x * blockDim.x + threadIdx.x;
  if (c >= C) return;
  float mx = -1e30f;
  for (int s = 0; s < nsample; ++s)
    mx = fmaxf(mx, (float)A[frag_addr(c, s, MtilesIn)]);
  outF[frag_addr(c, bcol, 1)] = (_Float16)mx;
}

// ---------------------------------------------------------------------------
// Final outputs: pred_box (B,7) then pred_class (B,) = sigmoid.
// Head outputs are B-frag with Mtiles=1: element (k=o, m=b) at b*16+o.
// ---------------------------------------------------------------------------
__global__ void finalize_kernel(const _Float16* __restrict__ Hd, const _Float16* __restrict__ Hc,
                                float* __restrict__ out)
{
  int t = threadIdx.x;
  if (t < 112) {
    int b = t / 7, o = t - b * 7;
    out[t] = (float)Hd[b * 16 + o];
  } else if (t < 128) {
    int b = t - 112;
    float v = (float)Hc[b * 16];
    out[112 + b] = 1.0f / (1.0f + expf(-v));
  }
}

// ---------------------------------------------------------------------------
// Host orchestration
// ---------------------------------------------------------------------------
extern "C" void kernel_launch(void* const* d_in, const int* in_sizes, int n_in,
                              void* d_out, int out_size, void* d_ws, size_t ws_size,
                              hipStream_t stream)
{
  const int kB = 16, kN = 4096;
  const float* x = (const float*)d_in[0];
  auto Wp = [&](int i) { return (const float*)d_in[i]; };

  char* base = (char*)d_ws;
  size_t off = 0;
  auto alloc = [&](size_t bytes) -> char* {
    char* p = base + off;
    off += (bytes + 255) & ~(size_t)255;
    return p;
  };
  auto cdiv = [](int a, int b) { return (a + b - 1) / b; };

  float*    xyz0  = (float*)   alloc((size_t)kB * kN * 3 * 4);
  _Float16* feat0 = (_Float16*)alloc((size_t)kB * kN * 2);
  int*      fidx1 = (int*)     alloc((size_t)kB * 1024 * 4);
  float*    nxyz1 = (float*)   alloc((size_t)kB * 1024 * 3 * 4);
  int*      idx1  = (int*)     alloc((size_t)kB * 1024 * 32 * 4);
  _Float16* feat1 = (_Float16*)alloc((size_t)kB * 64 * 1024 * 2);
  int*      fidx2 = (int*)     alloc((size_t)kB * 256 * 4);
  float*    nxyz2 = (float*)   alloc((size_t)kB * 256 * 3 * 4);
  int*      idx2  = (int*)     alloc((size_t)kB * 256 * 32 * 4);
  _Float16* feat2 = (_Float16*)alloc((size_t)kB * 128 * 256 * 2);
  int*      fidx3 = (int*)     alloc((size_t)kB * 64 * 4);
  float*    nxyz3 = (float*)   alloc((size_t)kB * 64 * 3 * 4);
  int*      idx3  = (int*)     alloc((size_t)kB * 64 * 32 * 4);
  _Float16* feat3 = (_Float16*)alloc((size_t)kB * 256 * 64 * 2);
  _Float16* featH = (_Float16*)alloc((size_t)512 * 16 * 2);
  _Float16* H1    = (_Float16*)alloc((size_t)256 * 16 * 2);
  _Float16* H2    = (_Float16*)alloc((size_t)256 * 16 * 2);
  _Float16* Hc3   = (_Float16*)alloc((size_t)32 * 16 * 2);
  _Float16* Hd3   = (_Float16*)alloc((size_t)32 * 16 * 2);
  _Float16* ping  = (_Float16*)alloc((size_t)64 * 32768 * 2);   // 4 MB
  _Float16* pong  = (_Float16*)alloc((size_t)64 * 32768 * 2);   // 4 MB

  // ---- weight pre-pack into A-fragment layout (18 layers) ----
  // {w d_in index, Cout, K, Kpad}
  const int LW[18][4] = {
      {1, 32, 4, 32},    {3, 32, 32, 32},   {5, 64, 32, 32},
      {7, 64, 67, 96},   {9, 64, 64, 64},   {11, 128, 64, 64},
      {13, 128, 131, 160}, {15, 128, 128, 128}, {17, 256, 128, 128},
      {19, 256, 259, 288}, {21, 256, 256, 256}, {23, 512, 256, 256},
      {25, 256, 512, 512}, {27, 256, 256, 256}, {29, 1, 256, 256},
      {31, 256, 512, 512}, {33, 256, 256, 256}, {35, 7, 256, 256}};
  _Float16* AP[18];
  for (int l = 0; l < 18; ++l) {
    int Cout = LW[l][1], K = LW[l][2], KC = LW[l][3] / 32;
    int total = cdiv(Cout, 16) * KC * 512;
    AP[l] = (_Float16*)alloc((size_t)total * 2);
    pack_w_kernel<<<cdiv(total, 256), 256, 0, stream>>>(Wp(LW[l][0]), AP[l], Cout, K, KC, total);
  }

  // ---- split x ----
  split_x_kernel<<<cdiv(kB * kN, 256), 256, 0, stream>>>(x, xyz0, feat0, kB * kN);

  // ---- SA levels 1..3 ----
  const int   Ln[3]       = {4096, 1024, 256};
  const int   Lnp[3]      = {1024, 256, 64};
  const float Lr2[3]      = {0.2f * 0.2f, 0.4f * 0.4f, 0.8f * 0.8f};
  const int   LC[3]       = {1, 64, 128};
  const int   LKpad[3]    = {32, 96, 160};
  const int   LCout[3][3] = {{32, 32, 64}, {64, 64, 128}, {128, 128, 256}};
  const float* Lxyz[3] = {xyz0, nxyz1, nxyz2};
  float*      Lnxyz[3] = {nxyz1, nxyz2, nxyz3};
  int*        Lfidx[3] = {fidx1, fidx2, fidx3};
  int*        Lidx[3]  = {idx1, idx2, idx3};
  _Float16*   Lfeat[3] = {feat0, feat1, feat2};
  _Float16*   Lpool[3] = {feat1, feat2, feat3};

  const int NS = 32;
  for (int L = 0; L < 3; ++L) {
    const int n = Ln[L], np = Lnp[L], C = LC[L], Kp = LKpad[L];
    const int M = np * NS, Mt = M / 16;

    size_t fps_lds = (size_t)n * 4 + FPS_BLOCK * 8;
    fps_kernel<<<kB, FPS_BLOCK, fps_lds, stream>>>(Lxyz[L], Lfidx[L], n, np);
    gather_xyz_kernel<<<cdiv(kB * np * 3, 256), 256, 0, stream>>>(
        Lxyz[L], Lfidx[L], Lnxyz[L], n, np, kB * np * 3);
    ballquery_kernel<<<(kB * np) / 8, 256, 0, stream>>>(
        Lxyz[L], Lnxyz[L], Lidx[L], n, np, Lr2[L], NS);

    for (int b = 0; b < kB; ++b) {
      group_kernel<<<cdiv(Kp * M, 256), 256, 0, stream>>>(
          Lxyz[L] + (size_t)b * n * 3, Lnxyz[L] + (size_t)b * np * 3,
          Lfeat[L] + (size_t)b * C * n, Lidx[L] + (size_t)b * np * NS,
          ping, n, NS, C, Mt, Kp * M);
      const _Float16* gin = ping;
      _Float16* gout = pong;
      int KC = Kp / 32;
      for (int l = 0; l < 3; ++l) {
        int Co = LCout[L][l];
        wmma_gemm_kernel<<<dim3(Mt, cdiv(Co, 16)), 32, 0, stream>>>(
            AP[L * 3 + l], Wp(2 + 2 * (L * 3 + l)), gin, gout, Co, KC, Mt, 1);
        const _Float16* t = gout; gout = (gout == pong) ? ping : pong;
        gin = t;
        KC = Co / 32;
      }
      int Cl = LCout[L][2];
      maxpool_kernel<<<cdiv(Cl * np, 256), 256, 0, stream>>>(
          gin, Lpool[L], Cl, np, NS, Mt, b * Cl * np, np);
    }
  }

  // ---- SA4 (global): grouped [288 x 64], MLP 259->256->256->512, max over 64 ----
  {
    const int M = 64, Mt = 4, C = 256, Kp = 288;
    const int Cout[3] = {256, 256, 512};
    for (int b = 0; b < kB; ++b) {
      group_all_kernel<<<cdiv(Kp * M, 256), 256, 0, stream>>>(
          nxyz3 + (size_t)b * 64 * 3, feat3 + (size_t)b * C * 64, ping, M, C, Mt, Kp * M);
      const _Float16* gin = ping;
      _Float16* gout = pong;
      int KC = Kp / 32;
      for (int l = 0; l < 3; ++l) {
        wmma_gemm_kernel<<<dim3(Mt, cdiv(Cout[l], 16)), 32, 0, stream>>>(
            AP[9 + l], Wp(2 + 2 * (9 + l)), gin, gout, Cout[l], KC, Mt, 1);
        const _Float16* t = gout; gout = (gout == pong) ? ping : pong;
        gin = t;
        KC = Cout[l] / 32;
      }
      maxpool_frag_kernel<<<cdiv(512, 256), 256, 0, stream>>>(gin, featH, 512, 64, Mt, b);
    }
  }

  // ---- heads: GEMMs with M = 16 (batch dim as columns), Mtiles = 1 ----
  wmma_gemm_kernel<<<dim3(1, 16), 32, 0, stream>>>(AP[12], Wp(26), featH, H1, 256, 16, 1, 1);
  wmma_gemm_kernel<<<dim3(1, 16), 32, 0, stream>>>(AP[13], Wp(28), H1, H2, 256, 8, 1, 1);
  wmma_gemm_kernel<<<dim3(1, 1),  32, 0, stream>>>(AP[14], Wp(30), H2, Hc3, 1, 8, 1, 0);
  wmma_gemm_kernel<<<dim3(1, 16), 32, 0, stream>>>(AP[15], Wp(32), featH, H1, 256, 16, 1, 1);
  wmma_gemm_kernel<<<dim3(1, 16), 32, 0, stream>>>(AP[16], Wp(34), H1, H2, 256, 8, 1, 1);
  wmma_gemm_kernel<<<dim3(1, 1),  32, 0, stream>>>(AP[17], Wp(36), H2, Hd3, 7, 8, 1, 0);

  finalize_kernel<<<1, 128, 0, stream>>>(Hd3, Hc3, (float*)d_out);
}
